// SharedMatrix_45079976739528
// MI455X (gfx1250) — matrix operand order repaired and checked by reading, not compiled, not run on hardware
//
#include <hip/hip_runtime.h>

// CDNA5 / gfx1250 butterfly block-sparse GEMM, double-buffered, 4x4 wave tiling:
//   out[:, ob*256:(ob+1)*256] = sum_{j<5} x[:, g_j*256:...] @ W[g_j*256:.., a_j, :]
// f32 -> bf16 (RNE via v_perm packing) staged through LDS in WMMA fragment order,
// conflict-free fragment layout, f32 accumulate in v_wmma_f32_16x16x32_bf16.
// All per-panel addressing reduced to (uniform base + thread-const) + immediate.

typedef __attribute__((ext_vector_type(16))) __bf16 v16bf;
typedef __attribute__((ext_vector_type(8)))  float  v8f;

#define IN_FEATURES  4096
#define OUT_FEATURES 4096
#define BS     256
#define ABPI   5
#define WROW   (ABPI * BS)           /* 1280 floats per weight row */
#define BM     128                   /* token tile per workgroup */
#define KP     32                    /* K panel per WMMA step */
#define NPANEL ((BS / KP) * ABPI)    /* 40 panels */

__device__ __forceinline__ unsigned int rne_bf16_hi(float f) {
    // round-to-nearest-even: bf16 result lives in bits [31:16]
    unsigned int u = __float_as_uint(f);
    return u + 0x7FFFu + ((u >> 16) & 1u);
}

__device__ __forceinline__ unsigned int pack_bf16(float lo, float hi) {
    // one v_perm_b32 merges the two rounded high halves: {hi[31:16], lo[31:16]}
    return __builtin_amdgcn_perm(rne_bf16_hi(hi), rne_bf16_hi(lo), 0x07060302u);
}

// Fragment tile = 1024B: elements h<8 at [lane*16 + h*2], h>=8 at [512 + lane*16 + (h-8)*2].
// Each ds_load_b128 (32 lanes x 16B = 512B) touches every bank exactly twice.
union FragU {
    uint4 q[2];
    v16bf v;
};

__device__ __forceinline__ v16bf ldfrag(const unsigned char* tile, int lane) {
    FragU f;
    f.q[0] = *(const uint4*)(tile + lane * 16);
    f.q[1] = *(const uint4*)(tile + 512 + lane * 16);
    return f.v;
}

// A: it-th load covers row = (tid>>3) + 32*it, k0 = (tid&7)*4  (immediate stride)
__device__ __forceinline__ void load_panel(const float* __restrict__ xbase,
                                           const float* __restrict__ wbase,
                                           int kk0, int tid,
                                           float4 (&ra)[4],
                                           float (&rb0)[16], float (&rb1)[16]) {
    const float* abase = xbase + kk0 + (tid >> 3) * IN_FEATURES + ((tid & 7) << 2);
    #pragma unroll
    for (int it = 0; it < 4; ++it)
        ra[it] = *(const float4*)(abase + (size_t)it * 32 * IN_FEATURES);

    // B: o = tid, k = 2*it  ->  one base, immediate offsets (2it)*WROW, (2it+1)*WROW
    const float* bbase = wbase + (size_t)kk0 * WROW + tid;
    #pragma unroll
    for (int it = 0; it < 16; ++it) {
        rb0[it] = bbase[(size_t)(2 * it) * WROW];
        rb1[it] = bbase[(size_t)(2 * it + 1) * WROW];
    }
}

__device__ __forceinline__ void store_panel(unsigned char* sA, unsigned char* sB,
                                            int tid,
                                            const float4 (&ra)[4],
                                            const float (&rb0)[16], const float (&rb1)[16]) {
    // A destination: thread-constant base, tile advances by 2 per it (+2048B)
    {
        const int row_t = tid >> 3;                   // 0..31
        const int k0    = (tid & 7) << 2;             // 0..28
        const int half  = (k0 >> 3) & 1;              // lane half selects K+8
        const int h     = ((k0 & 16) >> 1) + (k0 & 7);// 0,4,8,12
        const int ln    = (row_t & 15) + (half << 4);
        unsigned char* adst = sA + (row_t >> 4) * 1024 + ((h & 8) << 6)
                            + ln * 16 + (h & 7) * 2;
        #pragma unroll
        for (int it = 0; it < 4; ++it) {
            uint2 v;
            v.x = pack_bf16(ra[it].x, ra[it].y);
            v.y = pack_bf16(ra[it].z, ra[it].w);
            *(uint2*)(adst + it * 2048) = v;
        }
    }
    // B destination: o = tid (thread-constant), k = 2*it (immediate)
    {
        unsigned char* bdst = sB + (tid >> 4) * 1024 + (tid & 15) * 16;
        #pragma unroll
        for (int it = 0; it < 16; ++it) {
            const int k   = 2 * it;
            const int hb  = ((k & 16) >> 1) + (k & 7);
            const int off = (((it >> 2) & 1) << 8)    /* lane half: +16 lanes * 16B */
                          + ((hb & 8) << 6) + (hb & 7) * 2;
            *(unsigned int*)(bdst + off) = pack_bf16(rb0[it], rb1[it]);
        }
    }
}

// Wave tiling: 4 M-tiles x 4 N-tiles per wave; each B fragment feeds 4 WMMAs.
__device__ __forceinline__ void compute_panel(const unsigned char* sA,
                                              const unsigned char* sB,
                                              int mg, int ng, int lane, v8f (&acc)[16]) {
    v16bf af[4];
    #pragma unroll
    for (int a = 0; a < 4; ++a)
        af[a] = ldfrag(sA + (mg * 4 + a) * 1024, lane);
    #pragma unroll
    for (int b = 0; b < 4; ++b) {
        v16bf bf = ldfrag(sB + (ng * 4 + b) * 1024, lane);
        #pragma unroll
        for (int a = 0; a < 4; ++a) {
            acc[a * 4 + b] = __builtin_amdgcn_wmma_f32_16x16x32_bf16(
                false, af[a], false, bf, (short)0, acc[a * 4 + b], false, false);
        }
    }
}

__global__ __launch_bounds__(256) void butterfly_wmma_kernel(
    const float* __restrict__ x,
    const float* __restrict__ weight,
    const int*   __restrict__ flat,
    float*       __restrict__ out)
{
    __shared__ __align__(32) unsigned char sA[2][8  * 1024];
    __shared__ __align__(32) unsigned char sB[2][16 * 1024];

    const int tid   = threadIdx.x;
    const int lane  = tid & 31;
    const int wv    = tid >> 5;          // wave id 0..7
    const int mg    = wv & 1;            // M tile group (tiles mg*4 .. mg*4+3)
    const int ng    = wv >> 1;           // N tile group (tiles ng*4 .. ng*4+3)
    const int ob    = blockIdx.x;        // output block 0..15
    const int mbase = blockIdx.y * BM;   // token tile base

    v8f acc[16] = {};
    float4 ra[4];
    float  rb0[16], rb1[16];

    // panel -> (x base, w base, k offset) ; flat[f] = g*5 + a
    auto panel_bases = [&](int p, const float*& xb, const float*& wb, int& kk0) {
        int j = p >> 3;
        kk0   = (p & 7) * KP;
        int f = flat[ob * ABPI + j];
        int g = f / ABPI;
        int a = f - g * ABPI;
        wb = weight + (size_t)g * BS * WROW + a * BS;
        xb = x + (size_t)mbase * IN_FEATURES + g * BS;
    };

    // prologue: stage panel 0 into buffer 0
    {
        const float* xb; const float* wb; int kk0;
        panel_bases(0, xb, wb, kk0);
        load_panel(xb, wb, kk0, tid, ra, rb0, rb1);
        store_panel(sA[0], sB[0], tid, ra, rb0, rb1);
    }
    __syncthreads();

    for (int p = 0; p < NPANEL - 1; ++p) {
        const int cur = p & 1;
        {   // issue next panel's global loads before the matrix work
            const float* xb; const float* wb; int kk0;
            panel_bases(p + 1, xb, wb, kk0);
            load_panel(xb, wb, kk0, tid, ra, rb0, rb1);
        }
        compute_panel(sA[cur], sB[cur], mg, ng, lane, acc);   // hides load latency
        store_panel(sA[cur ^ 1], sB[cur ^ 1], tid, ra, rb0, rb1);
        __syncthreads();
    }
    compute_panel(sA[(NPANEL - 1) & 1], sB[(NPANEL - 1) & 1], mg, ng, lane, acc);

    // writeback: C/D layout lane = N, VGPR r = M row r (+8 for upper half-lanes).
    // Non-temporal: out is streamed once; keep x/weight resident in the 192MB L2.
    #pragma unroll
    for (int a = 0; a < 4; ++a) {
        const int mrow = mbase + (mg * 4 + a) * 16 + ((lane >> 4) << 3);
        #pragma unroll
        for (int b = 0; b < 4; ++b) {
            const int col = ob * BS + (ng * 4 + b) * 16 + (lane & 15);
            float* op = out + (size_t)mrow * OUT_FEATURES + col;
            #pragma unroll
            for (int r = 0; r < 8; ++r) {
                __builtin_nontemporal_store(acc[a * 4 + b][r],
                                            op + (size_t)r * OUT_FEATURES);
            }
        }
    }
}

extern "C" void kernel_launch(void* const* d_in, const int* in_sizes, int n_in,
                              void* d_out, int out_size, void* d_ws, size_t ws_size,
                              hipStream_t stream) {
    const float* x      = (const float*)d_in[0];   // [8192, 4096] f32
    const float* weight = (const float*)d_in[1];   // [4096, 5, 256] f32
    const int*   flat   = (const int*)d_in[2];     // [16, 5] i32
    float*       out    = (float*)d_out;           // [8192, 4096] f32

    dim3 grid(OUT_FEATURES / BS, 8192 / BM);       // (16, 64)
    butterfly_wmma_kernel<<<grid, 256, 0, stream>>>(x, weight, flat, out);
}
